// Attention_46402826666629
// MI455X (gfx1250) — matrix operand order repaired and checked by reading, not compiled, not run on hardware
//
#include <hip/hip_runtime.h>
#include <hip/hip_bf16.h>
#include <math.h>

#define D_MODEL 768
#define N_HEAD  12
#define N_BR    4
#define DH      64
#define HALF_DH 32
#define H_TOT   48
#define BATCH   2
#define SEQ     1024

#define SILU_SCALE 1.8137993642342178f   // pi / sqrt(3)
#define ATTNSCALE  0.125f                // DH^-0.5

typedef __bf16 bf16;
typedef __attribute__((ext_vector_type(16))) __bf16 v16bf;
typedef __attribute__((ext_vector_type(8)))  __bf16 v8bf;
typedef __attribute__((ext_vector_type(8)))  float  v8f;

// ---------------------------------------------------------------- helpers ---
__device__ __forceinline__ bf16 f2bf(float f) {
  unsigned u = __builtin_bit_cast(unsigned, f);
  u += 0x7FFFu + ((u >> 16) & 1u);               // round-to-nearest-even
  unsigned short h = (unsigned short)(u >> 16);
  return __builtin_bit_cast(bf16, h);
}

__device__ __forceinline__ v16bf cat8(v8bf lo, v8bf hi) {
  v16bf r;
#pragma unroll
  for (int i = 0; i < 8; ++i) { r[i] = lo[i]; r[8 + i] = hi[i]; }
  return r;
}

// A-operand (16x32, M x K), CDNA5 16-bit A layout. Row-major A, lda in elements.
// lane L: M = L%15? no: M = L&15; K halves offset by (L>=16 ? 8 : 0).
// Per lane: 2 x 16-byte contiguous loads.
__device__ __forceinline__ v16bf load_a_vec(const bf16* base, int lda, int lane) {
  int r = lane & 15, hi = lane >> 4;
  const bf16* row = base + (long)r * lda + hi * 8;
  v8bf lo  = *(const v8bf*)row;          // K = hi*8 .. hi*8+7
  v8bf hi8 = *(const v8bf*)(row + 16);   // K = 16+hi*8 .. 16+hi*8+7
  return cat8(lo, hi8);
}

// B-operand (32x16, K x N), contiguous in K. Element (k,n) at base + n*nStride + k.
// lane half hi: K = hi*8..hi*8+7 and 16+hi*8..16+hi*8+7 (matches A).  Two 16-byte loads per lane.
__device__ __forceinline__ v16bf load_b_vec(const bf16* base, long nStride, int lane) {
  int n = lane & 15, hi = lane >> 4;
  const bf16* p = base + (long)n * nStride + hi * 8;
  v8bf lo  = *(const v8bf*)p;
  v8bf hi8 = *(const v8bf*)(p + 16);
  return cat8(lo, hi8);
}

// ---------------------------------------------------------------- kernels ---
__global__ void f32_to_bf16_kernel(const float* __restrict__ in, bf16* __restrict__ out, long n) {
  long i = (long)blockIdx.x * blockDim.x + threadIdx.x;
  if (i < n) out[i] = f2bf(in[i]);
}

// V (b,t,h,d) f32 -> Vt (b,h,d,t) bf16 so P@V B-operand is K-contiguous.
__global__ void v_transpose_kernel(const float* __restrict__ Vf, bf16* __restrict__ Vt) {
  long i = (long)blockIdx.x * blockDim.x + threadIdx.x;
  const long total = (long)BATCH * SEQ * N_HEAD * DH;
  if (i >= total) return;
  int d = (int)(i & (DH - 1));
  int h = (int)((i >> 6) % N_HEAD);
  long t = (i >> 6) / N_HEAD % SEQ;
  long b = i / ((long)SEQ * N_HEAD * DH);
  Vt[(((long)b * N_HEAD + h) * DH + d) * SEQ + t] = f2bf(Vf[i]);
}

// Wo (br,c,d) f32 -> WoT (d, br*768+c) bf16 so out-proj B-operand is K-contiguous.
__global__ void wo_transpose_kernel(const float* __restrict__ Wo, bf16* __restrict__ WoT) {
  long i = (long)blockIdx.x * blockDim.x + threadIdx.x;
  const long total = (long)D_MODEL * N_BR * D_MODEL;
  if (i >= total) return;
  long kap = i % (N_BR * D_MODEL);
  long d   = i / (N_BR * D_MODEL);
  WoT[i] = f2bf(Wo[kap * D_MODEL + d]);
}

// base1/base2: normalized Chebyshev pairs T_{1+2i}, T_{2+2i}, i = 0..31
__global__ void cheb_kernel(float* __restrict__ base1, float* __restrict__ base2) {
  int t = blockIdx.x * blockDim.x + threadIdx.x;
  if (t >= SEQ) return;
  float x = 2.0f * (float)t / (float)(SEQ - 1) - 1.0f;
  float tm1 = 1.0f, tc = x;                       // T0, T1
  for (int n = 1; n <= 63; n += 2) {
    float tn1 = 2.0f * x * tc - tm1;              // T_{n+1}
    int i = (n - 1) >> 1;
    float r1 = tc, r2 = tn1;
    float nrm = sqrtf(r1 * r1 + r2 * r2 + 1e-8f);
    base1[t * HALF_DH + i] = r1 / nrm;
    base2[t * HALF_DH + i] = r2 / nrm;
    float tn2 = 2.0f * x * tn1 - tc;              // T_{n+2}
    tm1 = tn1; tc = tn2;
  }
}

__global__ void bias_mean_kernel(const float* __restrict__ bo, float* __restrict__ bm) {
  int d = blockIdx.x * blockDim.x + threadIdx.x;
  if (d < D_MODEL)
    bm[d] = 0.25f * (bo[d] + bo[D_MODEL + d] + bo[2 * D_MODEL + d] + bo[3 * D_MODEL + d]);
}

// C[M,N] = scale * (A @ B^T) + bias[n].  64x64 output per wave: 16 WMMAs per K-step
// against 16 b128 loads -> 1:1 wmma:load.  A row-major (lda), B (N,K) row-major (bNs),
// both contiguous in K.
__global__ __launch_bounds__(256) void gemm_bf16_kernel(
    const bf16* __restrict__ A, int lda,
    const bf16* __restrict__ B, long bNs,
    float* __restrict__ C, int ldc,
    const float* __restrict__ bias, float scale,
    int M, int N, int K) {
  int wid  = (blockIdx.x * blockDim.x + threadIdx.x) >> 5;
  int lane = threadIdx.x & 31;
  int tilesN = N >> 6;
  int tiles  = (M >> 6) * tilesN;
  if (wid >= tiles) return;
  int tm = wid / tilesN, tn = wid % tilesN;
  const bf16* Ab = A + (long)(tm * 64) * lda;
  const bf16* Bb = B + (long)(tn * 64) * bNs;
  v8f acc[16] = {};                                // [mi*4 + ni]
  for (int k = 0; k < K; k += 32) {
    v16bf a[4];
#pragma unroll
    for (int mi = 0; mi < 4; ++mi)
      a[mi] = load_a_vec(Ab + (long)(mi * 16) * lda + k, lda, lane);
#pragma unroll
    for (int ni = 0; ni < 4; ++ni) {
      v16bf bm = load_b_vec(Bb + (long)(ni * 16) * bNs + k, bNs, lane);
#pragma unroll
      for (int mi = 0; mi < 4; ++mi)
        acc[mi * 4 + ni] = __builtin_amdgcn_wmma_f32_16x16x32_bf16(
            false, a[mi], false, bm, (short)0, acc[mi * 4 + ni], false, false);
    }
  }
  int n16 = lane & 15, hi = lane >> 4;
#pragma unroll
  for (int ni = 0; ni < 4; ++ni) {
    int n = tn * 64 + ni * 16 + n16;
    float bv = bias ? bias[n] : 0.0f;
#pragma unroll
    for (int mi = 0; mi < 4; ++mi) {
#pragma unroll
      for (int r = 0; r < 8; ++r) {
        int m = tm * 64 + mi * 16 + r + hi * 8;
        C[(long)m * ldc + n] = acc[mi * 4 + ni][r] * scale + bv;
      }
    }
  }
}

// q += q @ S[h]  (S = A - A^T + diag(id_bias[h])), then Chebyshev rotation, write bf16.
__global__ void q_transform_kernel(const float* __restrict__ Qf,
                                   const float* __restrict__ Amat,
                                   const float* __restrict__ id_bias,
                                   const float* __restrict__ base1,
                                   const float* __restrict__ base2,
                                   bf16* __restrict__ Qbf) {
  __shared__ float q0[DH];
  __shared__ float q1[DH];
  int idx = blockIdx.x;
  int h = idx % H_TOT;
  int t = (idx / H_TOT) % SEQ;
  int b = idx / (H_TOT * SEQ);
  int e = threadIdx.x;                            // 0..63
  long rowoff = (((long)b * SEQ + t) * H_TOT + h) * DH;
  q0[e] = Qf[rowoff + e];
  __syncthreads();
  float acc = 0.0f;
#pragma unroll 8
  for (int d = 0; d < DH; ++d) {
    float s = Amat[d * DH + e] - Amat[e * DH + d];
    if (d == e) s += id_bias[h * DH + d];
    acc += q0[d] * s;
  }
  q1[e] = q0[e] + acc;
  __syncthreads();
  float out;
  if (e < HALF_DH) {
    float b1 = base1[t * HALF_DH + e], b2 = base2[t * HALF_DH + e];
    out = q1[e] * b1 - q1[e + HALF_DH] * b2;
  } else {
    int i = e - HALF_DH;
    float b1 = base1[t * HALF_DH + i], b2 = base2[t * HALF_DH + i];
    out = q1[i] * b2 + q1[e] * b1;
  }
  Qbf[rowoff + e] = f2bf(out);
}

// rotate K, compute denom = sqrt(max(|k|^2, 1e-6)), write bf16 K and f32 denom.
__global__ void k_transform_kernel(const float* __restrict__ Kf,
                                   const float* __restrict__ base1,
                                   const float* __restrict__ base2,
                                   bf16* __restrict__ Kbf,
                                   float* __restrict__ kden) {
  __shared__ float k0[DH];
  __shared__ float sq[DH];
  int idx = blockIdx.x;
  int h = idx % N_HEAD;
  int t = (idx / N_HEAD) % SEQ;
  int b = idx / (N_HEAD * SEQ);
  int e = threadIdx.x;
  long rowoff = (((long)b * SEQ + t) * N_HEAD + h) * DH;
  k0[e] = Kf[rowoff + e];
  __syncthreads();
  float out;
  if (e < HALF_DH) {
    float b1 = base1[t * HALF_DH + e], b2 = base2[t * HALF_DH + e];
    out = k0[e] * b1 - k0[e + HALF_DH] * b2;
  } else {
    int i = e - HALF_DH;
    float b1 = base1[t * HALF_DH + i], b2 = base2[t * HALF_DH + i];
    out = k0[i] * b2 + k0[e] * b1;
  }
  sq[e] = out * out;
  Kbf[rowoff + e] = f2bf(out);
  __syncthreads();
  if (e == 0) {
    float s = 0.0f;
    for (int d = 0; d < DH; ++d) s += sq[d];
    s = fmaxf(s, 1e-6f);
    kden[((long)b * N_HEAD + h) * SEQ + t] = sqrtf(s);
  }
}

// one wave per (b, h, 16-query tile); streaming causal attention with sink normalization.
__global__ __launch_bounds__(256) void attention_kernel(
    const bf16* __restrict__ Qbf, const bf16* __restrict__ Kbf,
    const bf16* __restrict__ Vt, const float* __restrict__ kden,
    const float* __restrict__ sinks, const float* __restrict__ v_nulls,
    bf16* __restrict__ ctx) {
  __shared__ __align__(32) bf16 plds[8][16 * 32];
  int wib  = threadIdx.x >> 5;
  int lane = threadIdx.x & 31;
  int wid  = blockIdx.x * 8 + wib;
  int qt = wid & 63;
  int h  = (wid >> 6) % H_TOT;
  int b  = wid / (64 * H_TOT);
  int h12 = h % N_HEAD;
  int br  = h / N_HEAD;
  int t0  = qt * 16;
  int r16 = lane & 15, hi = lane >> 4;

  float sink = sinks[h];
  v8f o[4] = {};
  float rs[8];
#pragma unroll
  for (int r = 0; r < 8; ++r) rs[r] = 0.0f;

  const bf16* Qbase = Qbf + (((long)b * SEQ + t0) * H_TOT + h) * DH;
  const bf16* Vhead = Vt + (((long)b * N_HEAD + h12) * DH) * SEQ;
  int nChunks = (t0 >> 5) + 1;
  for (int sc = 0; sc < nChunks; ++sc) {
    int sBase = sc * 32;
#pragma unroll
    for (int half = 0; half < 2; ++half) {
      int sOff = sBase + half * 16;
      v8f s_acc = {};
#pragma unroll
      for (int kk = 0; kk < 2; ++kk) {
        int d0 = kk * 32;
        v16bf a = load_a_vec(Qbase + d0, H_TOT * DH, lane);
        // scores: n = key index s (row stride 12*64), K = d contiguous
        const bf16* Bb = Kbf + (((long)b * SEQ + sOff) * N_HEAD + h12) * DH + d0;
        v16bf bm = load_b_vec(Bb, (long)N_HEAD * DH, lane);
        s_acc = __builtin_amdgcn_wmma_f32_16x16x32_bf16(false, a, false, bm, (short)0, s_acc, false, false);
      }
      int s = sOff + r16;
      float mul = ATTNSCALE / kden[((long)b * N_HEAD + h12) * SEQ + s];
#pragma unroll
      for (int r = 0; r < 8; ++r) {
        int t = t0 + r + hi * 8;
        float xin = s_acc[r] * mul;
        float w = (xin > 20.0f) ? xin : log1pf(__expf(xin));  // softplus
        w = w / (1.0f + __expf(-SILU_SCALE * w));             // w * sigmoid(SCALE*w)
        if (s > t) w = 0.0f;                                  // causal
        rs[r] += w;
        plds[wib][(r + hi * 8) * 32 + half * 16 + r16] = f2bf(w);
      }
    }
    asm volatile("s_wait_dscnt 0" ::: "memory");   // LDS stores visible before A-operand reload
    v16bf pa = load_a_vec(&plds[wib][0], 32, lane);
#pragma unroll
    for (int nc = 0; nc < 4; ++nc) {
      // P@V: n = d (row stride SEQ in Vt), K = s contiguous
      v16bf bv = load_b_vec(Vhead + (long)(nc * 16) * SEQ + sBase, SEQ, lane);
      o[nc] = __builtin_amdgcn_wmma_f32_16x16x32_bf16(false, pa, false, bv, (short)0, o[nc], false, false);
    }
  }
  // row sums: reduce across the 16 lanes holding the same M row
#pragma unroll
  for (int r = 0; r < 8; ++r) {
    float v = rs[r];
    v += __shfl_xor(v, 1, 16);
    v += __shfl_xor(v, 2, 16);
    v += __shfl_xor(v, 4, 16);
    v += __shfl_xor(v, 8, 16);
    rs[r] = 1.0f / (v + sink + 1e-6f);             // alpha
  }
#pragma unroll
  for (int nc = 0; nc < 4; ++nc) {
    int d = nc * 16 + r16;
    float vn = v_nulls[h * DH + d];
#pragma unroll
    for (int r = 0; r < 8; ++r) {
      int t = t0 + r + hi * 8;
      float val = o[nc][r] * rs[r] + sink * rs[r] * vn;
      ctx[(((long)b * SEQ + t) * N_BR + br) * D_MODEL + h12 * DH + d] = f2bf(val);
    }
  }
}

// ----------------------------------------------------------------- launch ---
extern "C" void kernel_launch(void* const* d_in, const int* in_sizes, int n_in,
                              void* d_out, int out_size, void* d_ws, size_t ws_size,
                              hipStream_t stream) {
  (void)in_sizes; (void)n_in; (void)out_size; (void)ws_size;
  const float* X       = (const float*)d_in[0];
  const float* Wq      = (const float*)d_in[1];
  const float* bq      = (const float*)d_in[2];
  const float* Wk      = (const float*)d_in[3];
  const float* bk      = (const float*)d_in[4];
  const float* Wv      = (const float*)d_in[5];
  const float* bv      = (const float*)d_in[6];
  const float* Amat    = (const float*)d_in[7];
  const float* id_bias = (const float*)d_in[8];
  const float* sinks   = (const float*)d_in[9];
  const float* v_nulls = (const float*)d_in[10];
  const float* Wo      = (const float*)d_in[11];
  const float* bo      = (const float*)d_in[12];
  float* out = (float*)d_out;

  char* p = (char*)d_ws;
  auto alloc = [&](size_t bytes) -> char* {
    char* r = p; p += (bytes + 255) & ~(size_t)255; return r;
  };
  const long BT = (long)BATCH * SEQ;               // 2048 tokens

  bf16*  Xbf   = (bf16*) alloc(BT * D_MODEL * 2);
  bf16*  Wqbf  = (bf16*) alloc((long)N_BR * D_MODEL * D_MODEL * 2);
  bf16*  Wkbf  = (bf16*) alloc((long)D_MODEL * D_MODEL * 2);
  bf16*  Wvbf  = (bf16*) alloc((long)D_MODEL * D_MODEL * 2);
  bf16*  WoT   = (bf16*) alloc((long)N_BR * D_MODEL * D_MODEL * 2);
  float* Qf    = (float*)alloc(BT * (long)N_BR * D_MODEL * 4);
  float* Kf    = (float*)alloc(BT * D_MODEL * 4);
  float* Vf    = (float*)alloc(BT * D_MODEL * 4);
  bf16*  Qbf   = (bf16*) alloc(BT * (long)N_BR * D_MODEL * 2);
  bf16*  Kbf   = (bf16*) alloc(BT * D_MODEL * 2);
  bf16*  Vtb   = (bf16*) alloc(BT * D_MODEL * 2);
  bf16*  Ctx   = (bf16*) alloc(BT * (long)N_BR * D_MODEL * 2);
  float* kden  = (float*)alloc((long)BATCH * N_HEAD * SEQ * 4);
  float* base1 = (float*)alloc((long)SEQ * HALF_DH * 4);
  float* base2 = (float*)alloc((long)SEQ * HALF_DH * 4);
  float* biasm = (float*)alloc(D_MODEL * 4);

  auto cdiv = [](long a, long b) -> unsigned { return (unsigned)((a + b - 1) / b); };
  auto cvt = [&](const float* src, bf16* dst, long n) {
    f32_to_bf16_kernel<<<cdiv(n, 256), 256, 0, stream>>>(src, dst, n);
  };

  cvt(X,  Xbf,  BT * D_MODEL);
  cvt(Wq, Wqbf, (long)N_BR * D_MODEL * D_MODEL);
  cvt(Wk, Wkbf, (long)D_MODEL * D_MODEL);
  cvt(Wv, Wvbf, (long)D_MODEL * D_MODEL);
  wo_transpose_kernel<<<cdiv((long)N_BR * D_MODEL * D_MODEL, 256), 256, 0, stream>>>(Wo, WoT);

  cheb_kernel<<<SEQ / 256, 256, 0, stream>>>(base1, base2);
  bias_mean_kernel<<<cdiv(D_MODEL, 256), 256, 0, stream>>>(bo, biasm);

  auto gemm = [&](const bf16* A, int lda, const bf16* B, long bNs,
                  float* C, int ldc, const float* bias, float scale,
                  int M, int N, int K) {
    long tiles = (long)(M / 64) * (N / 64);
    gemm_bf16_kernel<<<cdiv(tiles, 8), 256, 0, stream>>>(A, lda, B, bNs, C, ldc, bias, scale, M, N, K);
  };

  // Q = X @ Wq^T + bq ; K = X @ Wk^T + bk ; V = X @ Wv^T + bv
  gemm(Xbf, D_MODEL, Wqbf, D_MODEL, Qf, N_BR * D_MODEL, bq, 1.0f, (int)BT, N_BR * D_MODEL, D_MODEL);
  gemm(Xbf, D_MODEL, Wkbf, D_MODEL, Kf, D_MODEL,        bk, 1.0f, (int)BT, D_MODEL,        D_MODEL);
  gemm(Xbf, D_MODEL, Wvbf, D_MODEL, Vf, D_MODEL,        bv, 1.0f, (int)BT, D_MODEL,        D_MODEL);

  v_transpose_kernel<<<cdiv(BT * D_MODEL, 256), 256, 0, stream>>>(Vf, Vtb);
  q_transform_kernel<<<BATCH * SEQ * H_TOT,  64, 0, stream>>>(Qf, Amat, id_bias, base1, base2, Qbf);
  k_transform_kernel<<<BATCH * SEQ * N_HEAD, 64, 0, stream>>>(Kf, base1, base2, Kbf, kden);

  attention_kernel<<<(BATCH * H_TOT * 64) / 8, 256, 0, stream>>>(Qbf, Kbf, Vtb, kden, sinks, v_nulls, Ctx);

  // y = 0.25 * sum_br(ctx_br @ WoT^T) + 0.25 * sum_br bo_br   (K = br*768+c concatenated)
  gemm(Ctx, N_BR * D_MODEL, WoT, N_BR * D_MODEL, out, D_MODEL, biasm, 0.25f,
       (int)BT, D_MODEL, N_BR * D_MODEL);
}
